// KernelNN_2740189135776
// MI455X (gfx1250) — matrix operand order repaired and checked by reading, not compiled, not run on hardware
//
#include <hip/hip_runtime.h>
#include <hip/hip_bf16.h>

// ---------------------------------------------------------------------------
// Problem constants (match reference)
// ---------------------------------------------------------------------------
#define NN      50000
#define EE      400000
#define IN_W    6
#define WIDTH   32
#define KER_W   256
#define DEPTH   4

typedef __attribute__((ext_vector_type(16))) _Float16 v16h;
typedef __attribute__((ext_vector_type(8)))  _Float16 v8h;
typedef __attribute__((ext_vector_type(8)))  float    v8f;

union V16 {
    v16h v;
    v8h  h[2];
};

// ---------------------------------------------------------------------------
// Prep: transpose k2_w [256,256] and k3_w [256,1024] into f16 [N][K] layout
// ---------------------------------------------------------------------------
__global__ void prep_weights_kernel(const float* __restrict__ k2w,
                                    const float* __restrict__ k3w,
                                    _Float16* __restrict__ k2T,
                                    _Float16* __restrict__ k3T) {
    int idx = blockIdx.x * 256 + threadIdx.x;      // 0 .. 1024*256-1
    if (idx < 256 * 256) {
        int n = idx >> 8, k = idx & 255;
        k2T[idx] = (_Float16)k2w[k * 256 + n];
    }
    if (idx < 1024 * 256) {
        int n = idx >> 8, k = idx & 255;
        k3T[idx] = (_Float16)k3w[k * 1024 + n];
    }
}

// ---------------------------------------------------------------------------
// Edge MLP layer 1: h1 = relu(edge_attr @ k1_w + k1_b), f16 out.  K=6: VALU.
// ---------------------------------------------------------------------------
__global__ void layer1_kernel(const float* __restrict__ ea,
                              const float* __restrict__ k1w,
                              const float* __restrict__ k1b,
                              _Float16* __restrict__ h1) {
    long long idx = (long long)blockIdx.x * 256 + threadIdx.x;
    if (idx >= (long long)EE * KER_W) return;
    int c = (int)(idx & 255);
    long long e = idx >> 8;
    const float* a = ea + e * IN_W;
    float s = k1b[c];
#pragma unroll
    for (int i = 0; i < IN_W; ++i) s += a[i] * k1w[i * KER_W + c];
    h1[idx] = (_Float16)(s > 0.f ? s : 0.f);
}

// ---------------------------------------------------------------------------
// WMMA GEMM:  C[M,NCOLS] = act(A[M,256] @ B[256,NCOLS] + bias)
//   A: f16 row-major.  BT: f16 [NCOLS][256] (transposed B).
//   Block = 32 rows (2 M-tiles) staged in padded LDS; 8 waves sweep N-tiles.
//   Per N-tile: B fragments held in registers, reused across both M-tiles;
//   A fragments re-read from LDS per k-step (conflict-free ds_read_b128).
//   TRANS != 0: store C as per-edge transposed 32x32 (w_e layout [o][i]).
// ---------------------------------------------------------------------------
#define LDS_PAD 8   // pad halves per row: stride 528B -> conflict-free banks

template <int NCOLS, int RELU, int TRANS>
__global__ __launch_bounds__(256) void gemm_f16_wmma_kernel(
    const _Float16* __restrict__ A,
    const _Float16* __restrict__ BT,
    const float* __restrict__ bias,
    _Float16* __restrict__ C)
{
    __shared__ _Float16 ldsA[32][KER_W + LDS_PAD];     // ~16.5 KB
    const int tid  = threadIdx.x;
    const int lane = tid & 31;
    const int wave = tid >> 5;
    const long long m0 = (long long)blockIdx.x * 32;

    // Cooperative staging of A strip: 32x256 halves = 1024 chunks of 16B
    {
        const _Float16* gA = A + m0 * KER_W;
#pragma unroll
        for (int c = tid; c < 1024; c += 256) {
            int row = c >> 5;
            int col = (c & 31) * 8;
            *(v8h*)&ldsA[row][col] = *(const v8h*)&gA[row * KER_W + col];
        }
    }
    __syncthreads();

    // A fragment addressing (contiguous K, matching B):
    //   lanes 0-15 : row M=lane,    K = kk*32 + {0..15}
    //   lanes16-31 : row M=lane-16, K = kk*32 + {16..31}
    const int arow = lane & 15;
    const int ksel = (lane >> 4) * 16;
    // B fragment: lane l<16 holds column N=l, K contiguous {0..15};
    //             lane>=16 holds column N=l-16, K {16..31}
    const int bcol = lane & 15;
    const int bko  = (lane >> 4) * 16;

    constexpr int NTILES = NCOLS / 16;
#pragma unroll 1
    for (int nt = wave; nt < NTILES; nt += 8) {
        const _Float16* bbase = BT + (long long)(nt * 16 + bcol) * KER_W + bko;
        if (nt + 8 < NTILES)                       // prefetch next B tile
            __builtin_prefetch(bbase + 8 * 16 * KER_W, 0, 0);

        V16 bf[8];                                 // whole 16x256 B tile slice
#pragma unroll
        for (int kk = 0; kk < 8; ++kk)
            bf[kk].v = *(const v16h*)(bbase + kk * 32);

#pragma unroll
        for (int mt = 0; mt < 2; ++mt) {
            v8f acc = {};
#pragma unroll
            for (int kk = 0; kk < 8; ++kk) {
                V16 af;
                const _Float16* ap = &ldsA[mt * 16 + arow][kk * 32 + ksel];
                af.h[0] = *(const v8h*)ap;
                af.h[1] = *(const v8h*)(ap + 8);
                acc = __builtin_amdgcn_wmma_f32_16x16x32_f16(
                    false, af.v, false, bf[kk].v, (short)0, acc, false, false);
            }
            // C tile layout: VGPR r, lane<16: (M=r, N=lane);
            //                        lane>=16: (M=8+r, N=lane-16)
            const int n     = lane & 15;
            const int mhalf = (lane >> 4) * 8;
            const float bn  = bias[nt * 16 + n];
#pragma unroll
            for (int r = 0; r < 8; ++r) {
                float v = acc[r] + bn;
                if (RELU) v = v > 0.f ? v : 0.f;
                const long long m = m0 + mt * 16 + mhalf + r;
                if (!TRANS) {
                    C[m * NCOLS + (nt * 16 + n)] = (_Float16)v;
                } else {
                    int c = nt * 16 + n;         // c = i*32 + o in reference
                    C[m * 1024 + (long long)(c & 31) * 32 + (c >> 5)] = (_Float16)v;
                }
            }
        }
    }
}

// ---------------------------------------------------------------------------
// z0 = x @ fc1_w + fc1_b
// ---------------------------------------------------------------------------
__global__ void z0_kernel(const float* __restrict__ x,
                          const float* __restrict__ w,
                          const float* __restrict__ b,
                          float* __restrict__ z) {
    int idx = blockIdx.x * 256 + threadIdx.x;
    if (idx >= NN * WIDTH) return;
    int j = idx & 31;
    int n = idx >> 5;
    float s = b[j];
#pragma unroll
    for (int i = 0; i < IN_W; ++i) s += x[n * IN_W + i] * w[i * WIDTH + j];
    z[idx] = s;
}

__global__ void zero_kernel(float* __restrict__ p, long long n) {
    long long i = (long long)blockIdx.x * 256 + threadIdx.x;
    if (i < n) p[i] = 0.f;
}

__global__ void deg_kernel(const int* __restrict__ dst, float* __restrict__ deg) {
    int e = blockIdx.x * 256 + threadIdx.x;
    if (e < EE) atomicAdd(&deg[dst[e]], 1.0f);
}

__global__ void invdeg_kernel(float* __restrict__ deg) {
    int i = blockIdx.x * 256 + threadIdx.x;
    if (i < NN) deg[i] = 1.0f / fmaxf(deg[i], 1.0f);
}

// ---------------------------------------------------------------------------
// r = z @ root_w + conv_b
// ---------------------------------------------------------------------------
__global__ void root_kernel(const float* __restrict__ z,
                            const float* __restrict__ rw,
                            const float* __restrict__ cb,
                            float* __restrict__ r) {
    int idx = blockIdx.x * 256 + threadIdx.x;
    if (idx >= NN * WIDTH) return;
    int j = idx & 31;
    long long n = idx >> 5;
    const float* zr = z + n * WIDTH;
    float s = cb[j];
#pragma unroll
    for (int i = 0; i < WIDTH; ++i) s += zr[i] * rw[i * WIDTH + j];
    r[idx] = s;
}

// ---------------------------------------------------------------------------
// msg[e,o] = sum_i z[src[e], i] * w_e[e][o][i];  atomic scatter into agg[dst]
// One wave per edge; lane = output channel o; lane reads 64 contiguous bytes.
// ---------------------------------------------------------------------------
__global__ __launch_bounds__(256) void msg_kernel(const int* __restrict__ src,
                                                  const int* __restrict__ dst,
                                                  const float* __restrict__ z,
                                                  const _Float16* __restrict__ wt,
                                                  float* __restrict__ agg) {
    int lane = threadIdx.x & 31;
    long long e = (long long)blockIdx.x * 8 + (threadIdx.x >> 5);
    if (e >= EE) return;
    int s = src[e];
    int d = dst[e];
    const float* zs = z + (long long)s * WIDTH;
    const _Float16* wr = wt + e * 1024 + (long long)lane * 32;
    V16 w0, w1;
    w0.v = *(const v16h*)(wr);
    w1.v = *(const v16h*)(wr + 16);
    float m = 0.f;
#pragma unroll
    for (int i = 0; i < 16; ++i) m += zs[i] * (float)w0.v[i];
#pragma unroll
    for (int i = 0; i < 16; ++i) m += zs[16 + i] * (float)w1.v[i];
    atomicAdd(&agg[(long long)d * WIDTH + lane], m);
}

// ---------------------------------------------------------------------------
// z_next = relu(agg * inv_deg + r)
// ---------------------------------------------------------------------------
__global__ void update_kernel(const float* __restrict__ agg,
                              const float* __restrict__ inv,
                              const float* __restrict__ r,
                              float* __restrict__ zout) {
    int idx = blockIdx.x * 256 + threadIdx.x;
    if (idx >= NN * WIDTH) return;
    int n = idx >> 5;
    float v = agg[idx] * inv[n] + r[idx];
    zout[idx] = v > 0.f ? v : 0.f;
}

// ---------------------------------------------------------------------------
// out = z @ fc2_w + fc2_b
// ---------------------------------------------------------------------------
__global__ void fc2_kernel(const float* __restrict__ z,
                           const float* __restrict__ w,
                           const float* __restrict__ b,
                           float* __restrict__ out) {
    int n = blockIdx.x * 256 + threadIdx.x;
    if (n >= NN) return;
    float s = b[0];
#pragma unroll
    for (int j = 0; j < WIDTH; ++j) s += z[(long long)n * WIDTH + j] * w[j];
    out[n] = s;
}

// ---------------------------------------------------------------------------
extern "C" void kernel_launch(void* const* d_in, const int* in_sizes, int n_in,
                              void* d_out, int out_size, void* d_ws, size_t ws_size,
                              hipStream_t stream) {
    const float* x       = (const float*)d_in[0];
    const int*   ei      = (const int*)d_in[1];     // [2, E]
    const float* ea      = (const float*)d_in[2];
    const float* fc1_w   = (const float*)d_in[3];
    const float* fc1_b   = (const float*)d_in[4];
    const float* k1_w    = (const float*)d_in[5];
    const float* k1_b    = (const float*)d_in[6];
    const float* k2_w    = (const float*)d_in[7];
    const float* k2_b    = (const float*)d_in[8];
    const float* k3_w    = (const float*)d_in[9];
    const float* k3_b    = (const float*)d_in[10];
    const float* root_w  = (const float*)d_in[11];
    const float* conv_b  = (const float*)d_in[12];
    const float* fc2_w   = (const float*)d_in[13];
    const float* fc2_b   = (const float*)d_in[14];
    float* out = (float*)d_out;

    const int* src = ei;
    const int* dst = ei + EE;

    // -------- workspace carve-up (256B aligned) --------
    char* ws = (char*)d_ws;
    size_t off = 0;
    auto carve = [&](size_t bytes) -> void* {
        void* p = ws + off;
        off = (off + bytes + 255) & ~(size_t)255;
        return p;
    };
    _Float16* h1   = (_Float16*)carve((size_t)EE * KER_W * 2);   // 204.8 MB
    _Float16* h2   = (_Float16*)carve((size_t)EE * KER_W * 2);   // 204.8 MB
    _Float16* w_t  = (_Float16*)carve((size_t)EE * 1024 * 2);    // 819.2 MB
    _Float16* k2T  = (_Float16*)carve((size_t)256 * 256 * 2);
    _Float16* k3T  = (_Float16*)carve((size_t)1024 * 256 * 2);
    float*    zA   = (float*)carve((size_t)NN * WIDTH * 4);
    float*    zB   = (float*)carve((size_t)NN * WIDTH * 4);
    float*    rbuf = (float*)carve((size_t)NN * WIDTH * 4);
    float*    agg  = (float*)carve((size_t)NN * WIDTH * 4);
    float*    ideg = (float*)carve((size_t)NN * 4);
    (void)ws_size; (void)n_in; (void)in_sizes; (void)out_size;

    // -------- edge MLP (WMMA-heavy) --------
    prep_weights_kernel<<<1024, 256, 0, stream>>>(k2_w, k3_w, k2T, k3T);
    layer1_kernel<<<(int)(((long long)EE * KER_W + 255) / 256), 256, 0, stream>>>(ea, k1_w, k1_b, h1);
    gemm_f16_wmma_kernel<256, 1, 0><<<EE / 32, 256, 0, stream>>>(h1, k2T, k2_b, h2);
    gemm_f16_wmma_kernel<1024, 0, 1><<<EE / 32, 256, 0, stream>>>(h2, k3T, k3_b, w_t);

    // -------- degrees + z0 --------
    zero_kernel<<<(NN + 255) / 256, 256, 0, stream>>>(ideg, NN);
    deg_kernel<<<(EE + 255) / 256, 256, 0, stream>>>(dst, ideg);
    invdeg_kernel<<<(NN + 255) / 256, 256, 0, stream>>>(ideg);
    z0_kernel<<<(NN * WIDTH + 255) / 256, 256, 0, stream>>>(x, fc1_w, fc1_b, zA);

    // -------- DEPTH propagation steps (ping-pong z) --------
    float* zc = zA;
    float* zn = zB;
    for (int d = 0; d < DEPTH; ++d) {
        root_kernel<<<(NN * WIDTH + 255) / 256, 256, 0, stream>>>(zc, root_w, conv_b, rbuf);
        zero_kernel<<<(NN * WIDTH + 255) / 256, 256, 0, stream>>>(agg, (long long)NN * WIDTH);
        msg_kernel<<<EE / 8, 256, 0, stream>>>(src, dst, zc, w_t, agg);
        update_kernel<<<(NN * WIDTH + 255) / 256, 256, 0, stream>>>(agg, ideg, rbuf, zn);
        float* t = zc; zc = zn; zn = t;
    }

    // -------- head --------
    fc2_kernel<<<(NN + 255) / 256, 256, 0, stream>>>(zc, fc2_w, fc2_b, out);
}
